// Volume_35734127902876
// MI455X (gfx1250) — matrix operand order repaired and checked by reading, not compiled, not run on hardware
//
#include <hip/hip_runtime.h>

typedef __attribute__((ext_vector_type(16))) _Float16 v16h;
typedef __attribute__((ext_vector_type(8)))  _Float16 v8h;
typedef __attribute__((ext_vector_type(8)))  _Float16 h8;
typedef __attribute__((ext_vector_type(8)))  float    v8f;

#define WPB 8  // waves per 256-thread block (wave32)

static __device__ __forceinline__ void lds_fence() {
    // CDNA5 split dependency counter: wait for all outstanding DS ops so
    // cross-lane LDS producer->consumer within the wave is ordered.
    asm volatile("s_wait_dscnt 0" ::: "memory");
}

// f32-accumulating WMMA (density / color heads)
static __device__ __forceinline__ v8f wmma_f32(v16h a, v16h b, v8f c) {
    return __builtin_amdgcn_wmma_f32_16x16x32_f16(
        false, a, false, b, (short)0, c, false, false);
}

// f16-accumulating WMMA (hidden layers; D feeds f16 LDS staging directly)
static __device__ __forceinline__ v8h wmma_f16(v16h a, v16h b, v8h c) {
    return __builtin_amdgcn_wmma_f16_16x16x32_f16(
        false, a, false, b, (short)0, c, false, false);
}

// v_cvt_pk_rtz_f16_f32 packed convert, result as raw 32-bit payload
static __device__ __forceinline__ unsigned cvt_pk_u32(float x, float y) {
    return __builtin_bit_cast(unsigned, __builtin_amdgcn_cvt_pkrtz(x, y));
}

// Packed f16 relu on 4 WMMA accumulators in ONE asm block. Reading all four
// D vectors forces this block after all four WMMAs, so the first read of d0
// is separated from its producer by 3 multi-cycle XDL WMMAs and later reads
// by >=4 VALU ops -- satisfies the f16 WMMA->VALU RAW hazard (4 co-exec
// slots) that the compiler does not mitigate around inline asm.
static __device__ __forceinline__ void relu_4x8(v8h& x0, v8h& x1, v8h& x2, v8h& x3) {
    uint4 u0 = __builtin_bit_cast(uint4, x0);
    uint4 u1 = __builtin_bit_cast(uint4, x1);
    uint4 u2 = __builtin_bit_cast(uint4, x2);
    uint4 u3 = __builtin_bit_cast(uint4, x3);
    asm("v_pk_max_num_f16 %0, %0, 0\n\t"
        "v_pk_max_num_f16 %1, %1, 0\n\t"
        "v_pk_max_num_f16 %2, %2, 0\n\t"
        "v_pk_max_num_f16 %3, %3, 0\n\t"
        "v_pk_max_num_f16 %4, %4, 0\n\t"
        "v_pk_max_num_f16 %5, %5, 0\n\t"
        "v_pk_max_num_f16 %6, %6, 0\n\t"
        "v_pk_max_num_f16 %7, %7, 0\n\t"
        "v_pk_max_num_f16 %8, %8, 0\n\t"
        "v_pk_max_num_f16 %9, %9, 0\n\t"
        "v_pk_max_num_f16 %10, %10, 0\n\t"
        "v_pk_max_num_f16 %11, %11, 0\n\t"
        "v_pk_max_num_f16 %12, %12, 0\n\t"
        "v_pk_max_num_f16 %13, %13, 0\n\t"
        "v_pk_max_num_f16 %14, %14, 0\n\t"
        "v_pk_max_num_f16 %15, %15, 0"
        : "+v"(u0.x), "+v"(u0.y), "+v"(u0.z), "+v"(u0.w),
          "+v"(u1.x), "+v"(u1.y), "+v"(u1.z), "+v"(u1.w),
          "+v"(u2.x), "+v"(u2.y), "+v"(u2.z), "+v"(u2.w),
          "+v"(u3.x), "+v"(u3.y), "+v"(u3.z), "+v"(u3.w));
    x0 = __builtin_bit_cast(v8h, u0);
    x1 = __builtin_bit_cast(v8h, u1);
    x2 = __builtin_bit_cast(v8h, u2);
    x3 = __builtin_bit_cast(v8h, u3);
}

// Single-vector packed relu (used only on compiler-generated VALU results,
// which already sit >=1 VALU past the WMMA with compiler-managed hazards).
static __device__ __forceinline__ v8h relu8(v8h x) {
    uint4 u = __builtin_bit_cast(uint4, x);
    asm("v_pk_max_num_f16 %0, %1, 0" : "=v"(u.x) : "v"(u.x));
    asm("v_pk_max_num_f16 %0, %1, 0" : "=v"(u.y) : "v"(u.y));
    asm("v_pk_max_num_f16 %0, %1, 0" : "=v"(u.z) : "v"(u.z));
    asm("v_pk_max_num_f16 %0, %1, 0" : "=v"(u.w) : "v"(u.w));
    return __builtin_bit_cast(v8h, u);
}

// Build a B-operand (32x16 f16, K x N) register set from a row-major global
// weight matrix W[Krows x ncols]. Lane L holds column `ncol`, k = 16*(L/16)+e.
static __device__ __forceinline__ v16h load_B(const float* W, int ncols,
                                              int Krows, int Ncols,
                                              int ncol, int kbase, int group) {
    v16h b = {};
#pragma unroll
    for (int e = 0; e < 16; ++e) {
        int k = kbase + group * 16 + e;
        float v = 0.0f;
        if (k < Krows && ncol < Ncols) v = W[k * ncols + ncol];
        b[e] = (_Float16)v;
    }
    return b;
}

// Build an A-operand (16x32 f16) from row-major f16 LDS [16][rowStride].
// Lane L reads row L%16: two contiguous 16B runs (ds_load_b128 each).
static __device__ __forceinline__ v16h load_A(const _Float16* basep, int rowStride,
                                              int row, int kchunk, int group) {
    const int off = group * 16;
    const h8 r1 = *(const h8*)(basep + row * rowStride + kchunk + off);
    const h8 r2 = *(const h8*)(basep + row * rowStride + kchunk + off + 8);
    return __builtin_shufflevector(r1, r2, 0, 1, 2, 3, 4, 5, 6, 7,
                                   8, 9, 10, 11, 12, 13, 14, 15);
}

static __device__ __forceinline__ v8h splat8(_Float16 x) {
    v8h r;
#pragma unroll
    for (int e = 0; e < 8; ++e) r[e] = x;
    return r;
}

__global__ __launch_bounds__(256)
void nerf_mlp_wmma(const float* __restrict__ xyz,   // (P,3)
                   const float* __restrict__ ynm,   // (P,16)
                   const float* __restrict__ aabb,  // (2,3)
                   const float* __restrict__ W1,  const float* __restrict__ b1,
                   const float* __restrict__ W2,  const float* __restrict__ b2,
                   const float* __restrict__ Wd,  const float* __restrict__ bd,
                   const float* __restrict__ Wc1, const float* __restrict__ bc1,
                   const float* __restrict__ Wc2, const float* __restrict__ bc2,
                   float* __restrict__ outD,       // (P,)
                   float* __restrict__ outC,       // (P,3)
                   int numTiles) {
    const int tid   = threadIdx.x;
    const int lane  = tid & 31;
    const int w     = tid >> 5;       // wave index in block
    const int group = lane >> 4;      // 0: lanes 0-15, 1: lanes 16-31
    const int nl    = lane & 15;      // row/col-within-tile index

    __shared__ alignas(16) _Float16 sF[WPB][16][32];  // [f | ynm] staging
    __shared__ alignas(16) _Float16 sH[WPB][16][64];  // h / hc staging
    __shared__ float sMask[WPB][16];

    // ---- per-wave invariant state (weights in B layout, biases) ----
    // L1 B with bias folded: rows 0..2 = W1, row 3 = b1 (A supplies 1.0 at k=3).
    v16h B1[4];
#pragma unroll
    for (int t = 0; t < 4; ++t) {
        v16h b = {};
#pragma unroll
        for (int e = 0; e < 16; ++e) {
            int k = group * 16 + e;
            int n = 16 * t + nl;
            float v = 0.0f;
            if (k < 3)       v = W1[k * 64 + n];
            else if (k == 3) v = b1[n];
            b[e] = (_Float16)v;
        }
        B1[t] = b;
    }

    v16h B2[2], BC1[4], BC2[2];
#pragma unroll
    for (int c = 0; c < 2; ++c) B2[c]  = load_B(W2,  16, 64, 16, nl, 32 * c, group);
#pragma unroll
    for (int t = 0; t < 4; ++t) BC1[t] = load_B(Wc1, 64, 32, 64, 16 * t + nl, 0, group);
#pragma unroll
    for (int c = 0; c < 2; ++c) BC2[c] = load_B(Wc2, 3,  64, 3,  nl, 32 * c, group);

    // Fused density column: Wfuse = W2 @ Wd (64x1), bfuse = b2.Wd + bd.
    // Only column n==0 of this B-set is nonzero. (Rolled loops: once-per-wave
    // init, keep static code small.)
    v16h BD[2];
#pragma unroll
    for (int c = 0; c < 2; ++c) {
        v16h b = {};
        if (nl == 0) {
            for (int e = 0; e < 16; ++e) {
                int k = 32 * c + group * 16 + e;
                float acc = 0.0f;
                for (int j = 0; j < 16; ++j) acc += W2[k * 16 + j] * Wd[j];
                b[e] = (_Float16)acc;
            }
        }
        BD[c] = b;
    }
    float bfuse = bd[0];
    for (int j = 0; j < 16; ++j) bfuse += b2[j] * Wd[j];

    const v8h b2s = splat8((_Float16)b2[nl]);
    v8h bc1s[4];
#pragma unroll
    for (int t = 0; t < 4; ++t) bc1s[t] = splat8((_Float16)bc1[16 * t + nl]);
    const float bc2v = (nl < 3) ? bc2[nl] : 0.0f;

    const float a0 = aabb[0], a1c = aabb[1], a2c = aabb[2];
    const float s0 = 2.0f / (aabb[3] - a0);
    const float s1 = 2.0f / (aabb[4] - a1c);
    const float s2 = 2.0f / (aabb[5] - a2c);

    const int waveGlobal = blockIdx.x * (blockDim.x >> 5) + w;
    const int totalWaves = gridDim.x * (blockDim.x >> 5);

    const v8h zh = {};

    // A1 operand lives across iterations: elements 4..15 are structurally zero
    // (and elements 0..3 stay zero for lanes 16..31); only lanes 0-15 rewrite
    // elements 0..3 each tile.
    v16h A1 = {};

    for (int tile = waveGlobal; tile < numTiles; tile += totalWaves) {
        const int base = tile * 16;

        // prefetch next tile's inputs (global_prefetch_b8)
        if (tile + totalWaves < numTiles) {
            const int nb = (tile + totalWaves) * 16;
            __builtin_prefetch(ynm + (size_t)(nb + nl) * 16 + group * 8, 0, 0);
            if (group == 0) __builtin_prefetch(xyz + (size_t)(nb + nl) * 3, 0, 0);
        }

        // ---- stage ynm (cols 16..31 of sF): packed f32->f16 converts ----
        {
            const float* yp = ynm + (size_t)(base + nl) * 16 + group * 8;
            const float4 y0 = *(const float4*)yp;
            const float4 y1 = *(const float4*)(yp + 4);
            uint4 pkv;
            pkv.x = cvt_pk_u32(y0.x, y0.y);
            pkv.y = cvt_pk_u32(y0.z, y0.w);
            pkv.z = cvt_pk_u32(y1.x, y1.y);
            pkv.w = cvt_pk_u32(y1.z, y1.w);
            *(uint4*)(&sF[w][nl][16 + group * 8]) = pkv;
        }

        // ---- ndc + bounds mask + A1 operand (K padded 3->32) ----
        // k=3 carries the constant 1.0 that multiplies the bias row of B1.
        if (group == 0) {
            const float* xp = xyz + (size_t)(base + nl) * 3;
            const float nx = (xp[0] - a0)  * s0 - 1.0f;
            const float ny = (xp[1] - a1c) * s1 - 1.0f;
            const float nz = (xp[2] - a2c) * s2 - 1.0f;
            const bool inside = (nx >= -1.0f) && (nx <= 1.0f) &&
                                (ny >= -1.0f) && (ny <= 1.0f) &&
                                (nz >= -1.0f) && (nz <= 1.0f);
            sMask[w][nl] = inside ? 1.0f : 0.0f;
            A1[0] = (_Float16)nx; A1[1] = (_Float16)ny; A1[2] = (_Float16)nz;
            A1[3] = (_Float16)1.0f;
        }

        // ---- Layer 1: h = relu(ndc @ W1 + b1), f16 accumulate ----
        // f16 C/D layout: element e <-> row (e + 8*group), col nl.
        v8h d0 = wmma_f16(A1, B1[0], zh);
        v8h d1 = wmma_f16(A1, B1[1], zh);
        v8h d2 = wmma_f16(A1, B1[2], zh);
        v8h d3 = wmma_f16(A1, B1[3], zh);
        relu_4x8(d0, d1, d2, d3);
#pragma unroll
        for (int e = 0; e < 8; ++e) {
            const int m = e + 8 * group;
            sH[w][m][nl]      = d0[e];
            sH[w][m][16 + nl] = d1[e];
            sH[w][m][32 + nl] = d2[e];
            sH[w][m][48 + nl] = d3[e];
        }
        lds_fence();

        // ---- Layer 2 (features f16 + fused density column f32) ----
        const v16h a20 = load_A(&sH[w][0][0], 64, nl, 0,  group);
        const v16h a21 = load_A(&sH[w][0][0], 64, nl, 32, group);
        v8h f = zh;
        f = wmma_f16(a20, B2[0], f);
        f = wmma_f16(a21, B2[1], f);
        v8f dd = {};
        dd = wmma_f32(a20, BD[0], dd);
        dd = wmma_f32(a21, BD[1], dd);
        f = f + b2s;                               // v_pk_add_f16 (compiler-managed hazard)
#pragma unroll
        for (int e = 0; e < 8; ++e)
            sF[w][e + 8 * group][nl] = f[e];
        lds_fence();

        // ---- density = softplus(h @ Wfuse + bfuse), masked; lanes n==0 ----
        if (nl == 0) {
#pragma unroll
            for (int g = 0; g < 8; ++g) {
                const int m = g + 8 * group;
                const float x = dd[g] + bfuse;
                const float sp = (x > 20.0f) ? x : log1pf(__expf(x));
                outD[base + m] = sMask[w][m] * sp;
            }
        }

        // ---- A operand for [f | ynm] (16x32, K=32 exact) ----
        const v16h Af = load_A(&sF[w][0][0], 32, nl, 0, group);

        // ---- Color layer 1: hc = relu([f|ynm] @ Wc1 + bc1), f16 accumulate ----
        v8h e0 = wmma_f16(Af, BC1[0], zh);
        v8h e1 = wmma_f16(Af, BC1[1], zh);
        v8h e2 = wmma_f16(Af, BC1[2], zh);
        v8h e3 = wmma_f16(Af, BC1[3], zh);
        // bias add is compiler VALU (hazard-managed); asm relu reads add result
        e0 = relu8(e0 + bc1s[0]);
        e1 = relu8(e1 + bc1s[1]);
        e2 = relu8(e2 + bc1s[2]);
        e3 = relu8(e3 + bc1s[3]);
#pragma unroll
        for (int e = 0; e < 8; ++e) {
            const int m = e + 8 * group;
            sH[w][m][nl]      = e0[e];
            sH[w][m][16 + nl] = e1[e];
            sH[w][m][32 + nl] = e2[e];
            sH[w][m][48 + nl] = e3[e];
        }
        lds_fence();

        // ---- Color layer 2: sigmoid(hc @ Wc2 + bc2), N padded 3->16, f32 ----
        const v16h ac0 = load_A(&sH[w][0][0], 64, nl, 0,  group);
        const v16h ac1 = load_A(&sH[w][0][0], 64, nl, 32, group);
        v8f dc = {};
        dc = wmma_f32(ac0, BC2[0], dc);
        dc = wmma_f32(ac1, BC2[1], dc);
        if (nl < 3) {
#pragma unroll
            for (int g = 0; g < 8; ++g) {
                const int m = g + 8 * group;
                const float cv = 1.0f / (1.0f + __expf(-(dc[g] + bc2v)));
                outC[(size_t)(base + m) * 3 + nl] = sMask[w][m] * cv;
            }
        }
    }
}

extern "C" void kernel_launch(void* const* d_in, const int* in_sizes, int n_in,
                              void* d_out, int out_size, void* d_ws, size_t ws_size,
                              hipStream_t stream) {
    const float* xyz  = (const float*)d_in[0];
    const float* ynm  = (const float*)d_in[1];
    const float* aabb = (const float*)d_in[2];
    const float* W1   = (const float*)d_in[3];
    const float* b1   = (const float*)d_in[4];
    const float* W2   = (const float*)d_in[5];
    const float* b2   = (const float*)d_in[6];
    const float* Wd   = (const float*)d_in[7];
    const float* bd   = (const float*)d_in[8];
    const float* Wc1  = (const float*)d_in[9];
    const float* bc1  = (const float*)d_in[10];
    const float* Wc2  = (const float*)d_in[11];
    const float* bc2  = (const float*)d_in[12];

    const int NS = out_size / 4;          // density NS + color 3*NS
    float* outD = (float*)d_out;
    float* outC = (float*)d_out + NS;
    const int numTiles = NS / 16;

    const int blocks = 512;               // 512 blocks * 8 waves = 4096 waves
    nerf_mlp_wmma<<<blocks, 256, 0, stream>>>(
        xyz, ynm, aabb, W1, b1, W2, b2, Wd, bd, Wc1, bc1, Wc2, bc2,
        outD, outC, numTiles);
}
